// GATEncoder_24326694764708
// MI455X (gfx1250) — compile-verified, matrix operand order checked by reading, not run on hardware
//
#include <hip/hip_runtime.h>

#define WG 256

typedef __attribute__((ext_vector_type(16))) __bf16 v16bf;
typedef __attribute__((ext_vector_type(8)))  float  v8f;

__device__ __forceinline__ unsigned short f32_to_bf16(float f) {
    unsigned int u = __float_as_uint(f);
    unsigned int r = u + 0x7FFFu + ((u >> 16) & 1u);   // round to nearest even
    return (unsigned short)(r >> 16);
}

// ---------------- utility kernels ----------------

__global__ void k_fill_u32(unsigned int* __restrict__ p, long n, unsigned int v) {
    long i = (long)blockIdx.x * blockDim.x + threadIdx.x;
    if (i < n) p[i] = v;
}

__global__ void k_cvt_bf16(const float* __restrict__ in, unsigned short* __restrict__ out, long n) {
    long i = (long)blockIdx.x * blockDim.x + threadIdx.x;
    if (i < n) out[i] = f32_to_bf16(in[i]);
}

// wt[n*K + k] = bf16(w[k*Nout + n])   (weights are tiny; layout for per-lane contiguous B frags)
__global__ void k_cvt_transpose_bf16(const float* __restrict__ w, unsigned short* __restrict__ wt,
                                     int K, int Nout) {
    int i = blockIdx.x * blockDim.x + threadIdx.x;
    if (i >= K * Nout) return;
    int n = i / K;
    int k = i - n * K;
    wt[i] = f32_to_bf16(w[(size_t)k * Nout + n]);
}

// ---------------- WMMA GEMM: C[M,Nout] = A[M,K](bf16) x Bt[Nout,K](bf16)^T (+bias) ----------------
// one 16x16 tile per wave, 8 waves per block. M%16==0, K%32==0, Nout%16==0.

__global__ void k_wmma_gemm_bf16(const unsigned short* __restrict__ A,
                                 const unsigned short* __restrict__ Bt,
                                 float* __restrict__ C,
                                 const float* __restrict__ bias,
                                 int M, int K, int Nout) {
    int wave = threadIdx.x >> 5;
    int lane = threadIdx.x & 31;
    int ntn  = Nout >> 4;
    long tile = (long)blockIdx.x * 8 + wave;
    long total = ((long)(M >> 4)) * ntn;
    if (tile >= total) return;                  // uniform per wave -> EXEC all-1s inside

    int mt = (int)(tile / ntn);
    int nt = (int)(tile - (long)mt * ntn);
    int mbase = mt << 4;
    int nbase = nt << 4;

    int half = lane >> 4;        // 0: lanes 0-15, 1: lanes 16-31
    int l15  = lane & 15;

    const unsigned short* arow = A  + (size_t)(mbase + l15) * K;
    const unsigned short* brow = Bt + (size_t)(nbase + l15) * K;

    v8f acc = {};
    for (int k0 = 0; k0 < K; k0 += 32) {
        // 16-bit A fragment layout (ISA 7.12.2):
        //   lanes 0-15 : elems 0..7 = K k0..k0+7,   elems 8..15 = k0+16..k0+23
        //   lanes16-31 : elems 0..7 = K k0+8..k0+15, elems 8..15 = k0+24..k0+31
        int ka = k0 + half * 8;
        union { v16bf v; uint4 u[2]; } a, b;
        a.u[0] = *(const uint4*)(arow + ka);
        a.u[1] = *(const uint4*)(arow + ka + 16);
        b.u[0] = *(const uint4*)(brow + ka);
        b.u[1] = *(const uint4*)(brow + ka + 16);
        acc = __builtin_amdgcn_wmma_f32_16x16x32_bf16(false, a.v, false, b.v,
                                                      (short)0, acc, false, false);
    }

    // C/D layout: VGPR r -> row mbase + r + half*8 ; lane -> column nbase + l15
    int n  = nbase + l15;
    float bv = bias ? bias[n] : 0.0f;
    float* cp = C + (size_t)(mbase + half * 8) * Nout + n;
#pragma unroll
    for (int r = 0; r < 8; ++r)
        cp[(size_t)r * Nout] = acc[r] + bv;
}

// ---------------- attention coefficients ----------------
// asrc[n,h] = sum_c xh[n,h,c]*att_src[h,c] ; adst likewise. H==2.

__global__ void k_att_coef(const float* __restrict__ xh,
                           const float* __restrict__ atts, const float* __restrict__ attd,
                           float* __restrict__ asrc, float* __restrict__ adst,
                           int N, int ch) {
    int idx = blockIdx.x * blockDim.x + threadIdx.x;   // n*2+h
    if (idx >= N * 2) return;
    int n = idx >> 1, h = idx & 1;
    const float* xr = xh + (size_t)n * (2 * ch) + h * ch;
    const float* s  = atts + h * ch;
    const float* d  = attd + h * ch;
    float ss = 0.f, dd = 0.f;
    for (int c = 0; c < ch; ++c) { float v = xr[c]; ss += v * s[c]; dd += v * d[c]; }
    asrc[idx] = ss;
    adst[idx] = dd;
}

// ---------------- edge kernels (H==2 hardcoded) ----------------

__device__ __forceinline__ void edge_sd(const int* ei, int e, int E, int& s, int& d) {
    if (e < E) { s = ei[e]; d = ei[E + e]; }
    else       { s = d = e - E; }               // self loops appended
}

__global__ void k_edge_logits(const int* __restrict__ ei,
                              const float* __restrict__ asrc, const float* __restrict__ adst,
                              float* __restrict__ eb, unsigned int* __restrict__ mkey,
                              int E, int E2) {
    int idx = blockIdx.x * blockDim.x + threadIdx.x;   // e*2+h
    if (idx >= E2 * 2) return;
    int e = idx >> 1, h = idx & 1;
    int s, d; edge_sd(ei, e, E, s, d);
    float v = asrc[s * 2 + h] + adst[d * 2 + h];
    v = v > 0.f ? v : 0.2f * v;                        // leaky relu
    eb[idx] = v;
    unsigned int bits = __float_as_uint(v);
    unsigned int key  = (bits & 0x80000000u) ? ~bits : (bits | 0x80000000u);  // order-preserving
    atomicMax(&mkey[d * 2 + h], key);
}

__global__ void k_edge_exp(const int* __restrict__ ei,
                           float* __restrict__ eb, const unsigned int* __restrict__ mkey,
                           float* __restrict__ denom, int E, int E2) {
    int idx = blockIdx.x * blockDim.x + threadIdx.x;
    if (idx >= E2 * 2) return;
    int e = idx >> 1, h = idx & 1;
    int s, d; edge_sd(ei, e, E, s, d);
    unsigned int key  = mkey[d * 2 + h];
    unsigned int bits = (key & 0x80000000u) ? (key & 0x7FFFFFFFu) : ~key;
    float m  = __uint_as_float(bits);
    float ex = expf(eb[idx] - m);
    eb[idx] = ex;
    atomicAdd(&denom[d * 2 + h], ex);
}

// one thread = (edge, head, 4-channel chunk); cpt = ch/4 is a power of two
__global__ void k_edge_msg(const int* __restrict__ ei,
                           const float* __restrict__ eb, const float* __restrict__ denom,
                           const float* __restrict__ xh, float* __restrict__ out,
                           int E, int E2, int ch, int cshift) {
    long idx = (long)blockIdx.x * blockDim.x + threadIdx.x;
    int  cpt = ch >> 2;
    long tot = (long)E2 * 2 * cpt;
    if (idx >= tot) return;
    int  chunk = (int)(idx & (cpt - 1));
    long t = idx >> cshift;
    int  h = (int)(t & 1);
    int  e = (int)(t >> 1);
    int s, d; edge_sd(ei, e, E, s, d);
    float alpha = eb[(long)e * 2 + h] / (denom[d * 2 + h] + 1e-16f);
    const float4 xv = *(const float4*)(xh + (size_t)s * (2 * ch) + h * ch + chunk * 4);
    float* op = out + (size_t)d * (2 * ch) + h * ch + chunk * 4;
    atomicAdd(op + 0, alpha * xv.x);
    atomicAdd(op + 1, alpha * xv.y);
    atomicAdd(op + 2, alpha * xv.z);
    atomicAdd(op + 3, alpha * xv.w);
}

// ---------------- epilogues ----------------

__global__ void k_bias_relu(float* __restrict__ out, const float* __restrict__ bias,
                            int N, int C) {                 // C power of two
    int idx = blockIdx.x * blockDim.x + threadIdx.x;
    if (idx >= N * C) return;
    float v = out[idx] + bias[idx & (C - 1)];
    out[idx] = v > 0.f ? v : 0.f;
}

__global__ void k_col_stats(const float* __restrict__ x, float* __restrict__ cs,
                            float* __restrict__ cq, int N, int C) {
    __shared__ float s1[WG], s2[WG];
    int c = blockIdx.x;
    float a = 0.f, b = 0.f;
    for (int i = threadIdx.x; i < N; i += blockDim.x) {
        float v = x[(size_t)i * C + c];
        a += v; b += v * v;
    }
    s1[threadIdx.x] = a; s2[threadIdx.x] = b;
    __syncthreads();
    for (int s = WG / 2; s > 0; s >>= 1) {
        if ((int)threadIdx.x < s) {
            s1[threadIdx.x] += s1[threadIdx.x + s];
            s2[threadIdx.x] += s2[threadIdx.x + s];
        }
        __syncthreads();
    }
    if (threadIdx.x == 0) { cs[c] = s1[0]; cq[c] = s2[0]; }
}

__global__ void k_bn_apply_cvt(const float* __restrict__ x,
                               const float* __restrict__ cs, const float* __restrict__ cq,
                               const float* __restrict__ g, const float* __restrict__ be,
                               unsigned short* __restrict__ out,
                               int N, int C, float invN) {
    int idx = blockIdx.x * blockDim.x + threadIdx.x;
    if (idx >= N * C) return;
    int c = idx & (C - 1);
    float mu  = cs[c] * invN;
    float var = cq[c] * invN - mu * mu;
    float v = (x[idx] - mu) * rsqrtf(var + 1e-5f) * g[c] + be[c];
    out[idx] = f32_to_bf16(v);
}

// ---------------- host orchestration ----------------

extern "C" void kernel_launch(void* const* d_in, const int* in_sizes, int n_in,
                              void* d_out, int out_size, void* d_ws, size_t ws_size,
                              hipStream_t stream) {
    const float* x   = (const float*)d_in[0];
    const int*   ei  = (const int*)  d_in[1];
    const float* W1  = (const float*)d_in[2];
    const float* as1 = (const float*)d_in[3];
    const float* ad1 = (const float*)d_in[4];
    const float* b1  = (const float*)d_in[5];
    const float* g1  = (const float*)d_in[6];
    const float* be1 = (const float*)d_in[7];
    const float* W2  = (const float*)d_in[8];
    const float* as2 = (const float*)d_in[9];
    const float* ad2 = (const float*)d_in[10];
    const float* b2  = (const float*)d_in[11];
    const float* g2  = (const float*)d_in[12];
    const float* be2 = (const float*)d_in[13];
    const float* W3  = (const float*)d_in[14];
    const float* as3 = (const float*)d_in[15];
    const float* ad3 = (const float*)d_in[16];
    const float* b3  = (const float*)d_in[17];
    const float* lw  = (const float*)d_in[18];
    const float* lb  = (const float*)d_in[19];
    float* outp = (float*)d_out;

    const int N  = in_sizes[0] / 128;
    const int E  = in_sizes[1] / 2;
    const int E2 = E + N;

    // workspace carve-out
    char* ws = (char*)d_ws;
    size_t off = 0;
    auto ALLOC = [&](size_t bytes) -> char* {
        char* p = ws + off;
        off += (bytes + 255) & ~(size_t)255;
        return p;
    };
    unsigned short* Abf = (unsigned short*)ALLOC((size_t)N * 256 * 2); // bf16 GEMM input
    unsigned short* Wt  = (unsigned short*)ALLOC((size_t)256 * 256 * 2);
    float* XH    = (float*)       ALLOC((size_t)N * 256 * 4);
    float* OUT   = (float*)       ALLOC((size_t)N * 256 * 4);
    float* ASRC  = (float*)       ALLOC((size_t)N * 2 * 4);
    float* ADST  = (float*)       ALLOC((size_t)N * 2 * 4);
    float* EB    = (float*)       ALLOC((size_t)E2 * 2 * 4);
    unsigned int* MKEY = (unsigned int*)ALLOC((size_t)N * 2 * 4);
    float* DENOM = (float*)       ALLOC((size_t)N * 2 * 4);
    float* CS    = (float*)       ALLOC(256 * 4);
    float* CQ    = (float*)       ALLOC(256 * 4);

    auto cdiv = [](long a, long b) -> int { return (int)((a + b - 1) / b); };

    auto gat_layer = [&](const float* W, const float* atts, const float* attd,
                         const float* bias, int K, int ch) {
        int HC = 2 * ch;
        k_cvt_transpose_bf16<<<cdiv((long)K * HC, WG), WG, 0, stream>>>(W, Wt, K, HC);
        long tiles = ((long)N / 16) * (HC / 16);
        k_wmma_gemm_bf16<<<cdiv(tiles, 8), WG, 0, stream>>>(Abf, Wt, XH, nullptr, N, K, HC);
        k_att_coef<<<cdiv((long)N * 2, WG), WG, 0, stream>>>(XH, atts, attd, ASRC, ADST, N, ch);
        k_fill_u32<<<cdiv((long)N * 2, WG), WG, 0, stream>>>(MKEY, (long)N * 2, 0u);
        k_fill_u32<<<cdiv((long)N * 2, WG), WG, 0, stream>>>((unsigned int*)DENOM, (long)N * 2, 0u);
        k_fill_u32<<<cdiv((long)N * HC, WG), WG, 0, stream>>>((unsigned int*)OUT, (long)N * HC, 0u);
        k_edge_logits<<<cdiv((long)E2 * 2, WG), WG, 0, stream>>>(ei, ASRC, ADST, EB, MKEY, E, E2);
        k_edge_exp<<<cdiv((long)E2 * 2, WG), WG, 0, stream>>>(ei, EB, MKEY, DENOM, E, E2);
        int cpt = ch >> 2;
        int cshift = (cpt == 32) ? 5 : 4;
        long mt = (long)E2 * 2 * cpt;
        k_edge_msg<<<cdiv(mt, WG), WG, 0, stream>>>(ei, EB, DENOM, XH, OUT, E, E2, ch, cshift);
        k_bias_relu<<<cdiv((long)N * HC, WG), WG, 0, stream>>>(OUT, bias, N, HC);
    };

    // x -> bf16
    k_cvt_bf16<<<cdiv((long)N * 128, WG), WG, 0, stream>>>(x, Abf, (long)N * 128);

    // layer 1: K=128, ch=128, HC=256, then BN -> bf16
    gat_layer(W1, as1, ad1, b1, 128, 128);
    k_col_stats<<<256, WG, 0, stream>>>(OUT, CS, CQ, N, 256);
    k_bn_apply_cvt<<<cdiv((long)N * 256, WG), WG, 0, stream>>>(OUT, CS, CQ, g1, be1, Abf,
                                                               N, 256, 1.0f / (float)N);

    // layer 2: K=256, ch=64, HC=128, then BN -> bf16
    gat_layer(W2, as2, ad2, b2, 256, 64);
    k_col_stats<<<128, WG, 0, stream>>>(OUT, CS, CQ, N, 128);
    k_bn_apply_cvt<<<cdiv((long)N * 128, WG), WG, 0, stream>>>(OUT, CS, CQ, g2, be2, Abf,
                                                               N, 128, 1.0f / (float)N);

    // layer 3: K=128, ch=64, HC=128, no BN; convert relu output to bf16
    gat_layer(W3, as3, ad3, b3, 128, 64);
    k_cvt_bf16<<<cdiv((long)N * 128, WG), WG, 0, stream>>>(OUT, Abf, (long)N * 128);

    // final linear: [N,128] x [128,64] + lb -> d_out (f32)
    k_cvt_transpose_bf16<<<cdiv((long)128 * 64, WG), WG, 0, stream>>>(lw, Wt, 128, 64);
    long ftiles = ((long)N / 16) * (64 / 16);
    k_wmma_gemm_bf16<<<cdiv(ftiles, 8), WG, 0, stream>>>(Abf, Wt, outp, lb, N, 128, 64);
}
